// HeteroNodeEdgeAuxHead_7275674600380
// MI455X (gfx1250) — compile-verified, matrix operand order checked by reading, not run on hardware
//
#include <hip/hip_runtime.h>
#include <cstdint>

// CDNA5 (gfx1250) wave32 WMMA fragments
typedef __attribute__((ext_vector_type(16))) __bf16 v16bf;
typedef __attribute__((ext_vector_type(8)))  float  v8f;

#define DD 128   // node feature dim
#define HH 384   // 3*D edge MLP width

__device__ __forceinline__ unsigned short f2bf(float f) {
  unsigned int u = __float_as_uint(f);
  u += 0x7FFFu + ((u >> 16) & 1u);   // round-to-nearest-even
  return (unsigned short)(u >> 16);
}

__device__ __forceinline__ unsigned lds_off_of(const void* p) {
  // LDS aperture maps the LDS byte offset into addr[31:0]
  return (unsigned)(unsigned long long)(uintptr_t)p;
}

// ---- one-time converters -------------------------------------------------

__global__ void cvt_pairs_kernel(const float* __restrict__ in,
                                 unsigned short* __restrict__ out,
                                 long npairs) {
  long i = (long)blockIdx.x * blockDim.x + threadIdx.x;
  if (i < npairs) {
    float2 f = ((const float2*)in)[i];
    unsigned int p = (unsigned int)f2bf(f.x) | ((unsigned int)f2bf(f.y) << 16);
    ((unsigned int*)out)[i] = p;
  }
}

// out[n*K + k] = bf16(in[k*Nc + n])  (weights N-major: contiguous 32B/lane B-frags)
__global__ void cvt_transpose_kernel(const float* __restrict__ in,
                                     unsigned short* __restrict__ out,
                                     int K, int Nc) {
  int i = blockIdx.x * blockDim.x + threadIdx.x;
  if (i < K * Nc) {
    int n = i / K, k = i - n * K;
    out[i] = f2bf(in[(size_t)k * Nc + n]);
  }
}

__global__ void zero_f32_kernel(float* __restrict__ p, int n) {
  int i = blockIdx.x * blockDim.x + threadIdx.x;
  if (i < n) p[i] = 0.0f;
}

// ---- edge MLP: scores = relu([x[s]|x[d]|ea] @ V1 + bv1) @ V2 + bv2 -------
// one block = 32 edges (2 M-subtiles); 8 waves x 3 N-tiles cover H=384;
// each K-step: 3 B-fragments reused across 2 M-subtiles -> 6 WMMAs.

__global__ __launch_bounds__(256) void edge_mlp_kernel(
    const unsigned short* __restrict__ xbf,   // N x 128 bf16
    const float* __restrict__ edge_attr,      // E x 128 f32
    const unsigned short* __restrict__ V1T,   // 384 x 384 bf16, N-major
    const float* __restrict__ bv1,            // 384
    const float* __restrict__ V2,             // 384
    const float* __restrict__ bv2,            // 1
    const int* __restrict__ edge_index,       // [2][E]
    float* __restrict__ signal,               // N (pre-zeroed)
    int E_) {
  __shared__ __align__(32) unsigned short smA[32 * HH];  // 24 KB
  __shared__ float rowsum[32];
  __shared__ int   srcs[32];
  __shared__ int   dsts[32];

  const int tid  = threadIdx.x;
  const int wave = tid >> 5;
  const int lane = tid & 31;
  const int e0   = blockIdx.x << 5;

  if (tid < 32) {
    int e = e0 + tid;
    srcs[tid]   = (e < E_) ? edge_index[e]      : 0;
    dsts[tid]   = (e < E_) ? edge_index[E_ + e] : 0;
    rowsum[tid] = 0.0f;
  }
  __syncthreads();

  // --- stage x[src] | x[dst] (already bf16) via async copy to LDS --------
  // 32 rows x 2 segments x 8 x 16B chunks = 512 b128 transfers (ASYNCcnt).
  for (int j = tid; j < 512; j += 256) {
    const int row = j >> 4;
    const int c   = j & 15;
    const int seg = c >> 3;        // 0 = src, 1 = dst
    const int cc  = c & 7;         // 16B chunk within the 256B segment
    const int e   = e0 + row;
    unsigned short* lp = smA + (size_t)row * HH + seg * DD + cc * 8;
    if (e < E_) {
      const int node = seg ? dsts[row] : srcs[row];
      const unsigned short* gp = xbf + (size_t)node * DD + cc * 8;
      unsigned      lofs = lds_off_of(lp);
      unsigned long long ga = (unsigned long long)(uintptr_t)gp;
      asm volatile("global_load_async_to_lds_b128 %0, %1, off"
                   :: "v"(lofs), "v"(ga) : "memory");
    } else {
      *(uint4*)lp = make_uint4(0u, 0u, 0u, 0u);
    }
  }

  // --- stage edge_attr (f32 -> bf16 convert) through VGPRs ---------------
  for (int j = tid; j < 32 * (DD / 2); j += 256) {
    const int row = j >> 6;
    const int cp  = j & 63;
    const int e   = e0 + row;
    unsigned int packed = 0u;
    if (e < E_) {
      const float* ea = edge_attr + (size_t)e * DD + cp * 2;
      packed = (unsigned int)f2bf(ea[0]) | ((unsigned int)f2bf(ea[1]) << 16);
    }
    *(unsigned int*)(smA + (size_t)row * HH + 2 * DD + cp * 2) = packed;
  }

  asm volatile("s_wait_asynccnt 0x0" ::: "memory");  // my async copies landed
  __syncthreads();                                   // everyone's landed

  const int mrow  = lane & 15;
  const int half  = lane >> 4;
  const int nbase = wave * 48;   // this wave owns hidden cols [nbase, nbase+48)

  v8f a00 = {}, a01 = {}, a02 = {};   // M-subtile 0, N-tiles 0..2
  v8f a10 = {}, a11 = {}, a12 = {};   // M-subtile 1, N-tiles 0..2
  #pragma unroll 2
  for (int kk = 0; kk < 12; ++kk) {
    const int k = kk * 32 + (half << 4);
    v16bf f0 = *(const v16bf*)(smA + mrow * HH + k);          // rows 0..15
    v16bf f1 = *(const v16bf*)(smA + (16 + mrow) * HH + k);   // rows 16..31
    const unsigned short* bp = V1T + k;
    v16bf b0 = *(const v16bf*)(bp + (size_t)(nbase      + mrow) * HH);
    v16bf b1 = *(const v16bf*)(bp + (size_t)(nbase + 16 + mrow) * HH);
    v16bf b2 = *(const v16bf*)(bp + (size_t)(nbase + 32 + mrow) * HH);
    if (kk + 1 < 12)
      __builtin_prefetch(bp + 32 + (size_t)(nbase + mrow) * HH, 0, 1);
    a00 = __builtin_amdgcn_wmma_f32_16x16x32_bf16(false, f0, false, b0, (short)0, a00, false, false);
    a10 = __builtin_amdgcn_wmma_f32_16x16x32_bf16(false, f1, false, b0, (short)0, a10, false, false);
    a01 = __builtin_amdgcn_wmma_f32_16x16x32_bf16(false, f0, false, b1, (short)0, a01, false, false);
    a11 = __builtin_amdgcn_wmma_f32_16x16x32_bf16(false, f1, false, b1, (short)0, a11, false, false);
    a02 = __builtin_amdgcn_wmma_f32_16x16x32_bf16(false, f0, false, b2, (short)0, a02, false, false);
    a12 = __builtin_amdgcn_wmma_f32_16x16x32_bf16(false, f1, false, b2, (short)0, a12, false, false);
  }

  // Fused epilogue: +bv1, ReLU, dot with V2 slice, reduce per edge-row.
  // C-layout: VGPR r of lane l holds (M = r + 8*(l>>4), N = ntile*16 + (l&15)).
  const int n0 = nbase + mrow, n1 = n0 + 16, n2 = n0 + 32;
  const float bb0 = bv1[n0], bb1 = bv1[n1], bb2 = bv1[n2];
  const float w0 = V2[n0],  w1 = V2[n1],  w2 = V2[n2];
  #pragma unroll
  for (int r = 0; r < 8; ++r) {
    float s0 = fmaxf(a00[r] + bb0, 0.0f) * w0
             + fmaxf(a01[r] + bb1, 0.0f) * w1
             + fmaxf(a02[r] + bb2, 0.0f) * w2;
    float s1 = fmaxf(a10[r] + bb0, 0.0f) * w0
             + fmaxf(a11[r] + bb1, 0.0f) * w1
             + fmaxf(a12[r] + bb2, 0.0f) * w2;
    atomicAdd(&rowsum[half * 8 + r], s0);        // ds_add_f32
    atomicAdd(&rowsum[16 + half * 8 + r], s1);
  }
  __syncthreads();

  if (tid < 32) {
    int e = e0 + tid;
    if (e < E_) {
      float sc  = rowsum[tid] + bv2[0];
      float sig = 1.0f / (1.0f + __expf(-sc));     // in (0,1): positive
      // segment_max via int-bits atomicMax (valid for positive floats)
      atomicMax((int*)(signal + srcs[tid]), __float_as_int(sig));
    }
  }
}

// ---- node head: pred = relu(x@W1+b1)@W2 + b2 (+ ecw*signal on col 1) -----

__global__ __launch_bounds__(256) void node_head_kernel(
    const unsigned short* __restrict__ xbf,   // N x 128 bf16
    const unsigned short* __restrict__ W1T,   // 128 x 128 bf16, N-major
    const float* __restrict__ b1,             // 128
    const float* __restrict__ W2,             // 128 x 2
    const float* __restrict__ b2,             // 2
    const float* __restrict__ ecw,            // 1
    const float* __restrict__ signal,         // N
    float* __restrict__ out,                  // bs x 2
    int N_, int bs) {
  __shared__ float osum[16][2];
  const int tid  = threadIdx.x;
  const int wave = tid >> 5;
  const int lane = tid & 31;
  const int mrow = lane & 15;
  const int half = lane >> 4;
  const int m0   = blockIdx.x << 4;

  if (tid < 32) osum[tid >> 1][tid & 1] = 0.0f;
  __syncthreads();

  const int nbase = wave << 4;   // wave owns hidden cols [nbase, nbase+16)
  int row = m0 + mrow;
  if (row >= N_) row = N_ - 1;

  v8f acc = {};
  #pragma unroll
  for (int kk = 0; kk < 4; ++kk) {
    const int k = kk * 32 + (half << 4);
    v16bf a = *(const v16bf*)(xbf + (size_t)row * DD + k);
    v16bf b = *(const v16bf*)(W1T + (size_t)(nbase + mrow) * DD + k);
    acc = __builtin_amdgcn_wmma_f32_16x16x32_bf16(false, a, false, b, (short)0, acc, false, false);
  }

  const int n   = nbase + mrow;
  const float bb  = b1[n];
  const float w20 = W2[n * 2], w21 = W2[n * 2 + 1];
  #pragma unroll
  for (int r = 0; r < 8; ++r) {
    float hv = fmaxf(acc[r] + bb, 0.0f);
    atomicAdd(&osum[half * 8 + r][0], hv * w20);
    atomicAdd(&osum[half * 8 + r][1], hv * w21);
  }
  __syncthreads();

  if (tid < 32) {
    int r16 = tid >> 1, o = tid & 1;
    int grow = m0 + r16;
    if (grow < bs) {
      float v = osum[r16][o] + b2[o];
      if (o == 1) v += ecw[0] * signal[grow];
      out[(size_t)grow * 2 + o] = v;
    }
  }
}

__global__ void label_kernel(const int* __restrict__ y,
                             float* __restrict__ out, int bs) {
  int i = blockIdx.x * blockDim.x + threadIdx.x;
  if (i < bs) out[i] = (float)y[i];
}

// ---- launch --------------------------------------------------------------

extern "C" void kernel_launch(void* const* d_in, const int* in_sizes, int n_in,
                              void* d_out, int out_size, void* d_ws, size_t ws_size,
                              hipStream_t stream) {
  const float* x         = (const float*)d_in[0];
  const float* edge_attr = (const float*)d_in[1];
  const float* W1        = (const float*)d_in[2];
  const float* b1        = (const float*)d_in[3];
  const float* W2        = (const float*)d_in[4];
  const float* b2        = (const float*)d_in[5];
  const float* V1        = (const float*)d_in[6];
  const float* bv1       = (const float*)d_in[7];
  const float* V2        = (const float*)d_in[8];
  const float* bv2       = (const float*)d_in[9];
  const float* ecw       = (const float*)d_in[10];
  const int*   edge_index= (const int*)d_in[11];
  const int*   y         = (const int*)d_in[12];

  const int N_ = in_sizes[0] / DD;
  const int E_ = in_sizes[1] / DD;
  const int bs = out_size / 3;          // pred(bs*2) + label(bs)
  float* out = (float*)d_out;

  // workspace carve (256B aligned regions)
  char* ws = (char*)d_ws;
  size_t off = 0;
  auto carve = [&](size_t bytes) -> char* {
    char* p = ws + off;
    off = (off + bytes + 255) & ~(size_t)255;
    return p;
  };
  unsigned short* xbf    = (unsigned short*)carve((size_t)N_ * DD * 2);
  unsigned short* V1T    = (unsigned short*)carve((size_t)HH * HH * 2);
  unsigned short* W1T    = (unsigned short*)carve((size_t)DD * DD * 2);
  float*          signal = (float*)carve((size_t)N_ * 4);
  (void)ws_size; (void)n_in;

  long xpairs = (long)N_ * (DD / 2);
  cvt_pairs_kernel<<<(int)((xpairs + 255) / 256), 256, 0, stream>>>(x, xbf, xpairs);
  cvt_transpose_kernel<<<(HH * HH + 255) / 256, 256, 0, stream>>>(V1, V1T, HH, HH);
  cvt_transpose_kernel<<<(DD * DD + 255) / 256, 256, 0, stream>>>(W1, W1T, DD, DD);
  zero_f32_kernel<<<(N_ + 255) / 256, 256, 0, stream>>>(signal, N_);

  edge_mlp_kernel<<<(E_ + 31) / 32, 256, 0, stream>>>(
      xbf, edge_attr, V1T, bv1, V2, bv2, edge_index, signal, E_);

  node_head_kernel<<<(N_ + 15) / 16, 256, 0, stream>>>(
      xbf, W1T, b1, W2, b2, ecw, signal, out, N_, bs);

  label_kernel<<<(bs + 255) / 256, 256, 0, stream>>>(y, out + (size_t)2 * bs, bs);
}
